// PetaloMixer_69922067579577
// MI455X (gfx1250) — matrix operand order repaired and checked by reading, not compiled, not run on hardware
//
#include <hip/hip_runtime.h>

typedef __attribute__((ext_vector_type(16))) __bf16 v16bf;
typedef __attribute__((ext_vector_type(8)))  float  v8f;

#define BB 2
#define CC 256
#define LL 4096
#define NP 8          // 4*B mamba batch
#define DM 64         // Cq = d_model per part
#define DI 128        // d_inner
#define DS 16         // d_state
#define NCHUNK 16
#define CLEN 256      // LL / NCHUNK
#define MROWS (NP*LL) // 32768

static __device__ __forceinline__ unsigned short f2bf(float f) {
  unsigned u = __float_as_uint(f);
  u += 0x7FFFu + ((u >> 16) & 1u);   // round-to-nearest-even
  return (unsigned short)(u >> 16);
}
static __device__ __forceinline__ float silu_f(float x) { return x / (1.f + __expf(-x)); }
static __device__ __forceinline__ float softplus_f(float x) { return x > 20.f ? x : log1pf(__expf(x)); }

union FragU { v16bf v; unsigned u[8]; };

// A fragment: 16(M) x 32(K) bf16, row-major source [M][K].
// lanes 0-15: M=lane, K = {0..7} in u[0..3], {16..23} in u[4..7]
// lanes 16-31: M=lane-16, K = {8..15} and {24..31}
static __device__ __forceinline__ v16bf load_frag_a(const unsigned* A, int m0, int kb, int K, int lane) {
  int m = m0 + (lane & 15);
  int hi = lane >> 4;
  const unsigned* row = A + (((size_t)m * K) >> 1);
  FragU f;
  int p0 = (kb + hi * 8) >> 1;
  int p1 = (kb + 16 + hi * 8) >> 1;
#pragma unroll
  for (int j = 0; j < 4; ++j) { f.u[j] = row[p0 + j]; f.u[4 + j] = row[p1 + j]; }
  return f.v;
}

// B fragment: 32(K) x 16(N); element (k,n) = W[e0+n][k] with W row-major [E][K].
static __device__ __forceinline__ v16bf load_frag_b(const unsigned* W, int e0, int kb, int K, int lane) {
  int n = lane & 15;
  int kh = lane >> 4;
  const unsigned* row = W + (((size_t)(e0 + n) * K) >> 1);
  FragU f;
  int p0 = (kb + kh * 8) >> 1;
  int p1 = (kb + 16 + kh * 8) >> 1;
#pragma unroll
  for (int j = 0; j < 4; ++j) { f.u[j] = row[p0 + j]; f.u[4 + j] = row[p1 + j]; }
  return f.v;
}

// ---------------- weight repack: f32 -> bf16, A = -exp(A_log) ----------------
__global__ void k_repack(const float* in_w, const float* out_w, const float* proj_w,
                         const float* A_log, unsigned short* wb_in, unsigned short* wb_out,
                         unsigned short* wb_proj, float* Aneg) {
  int t = blockIdx.x * blockDim.x + threadIdx.x;
  if (t < 256 * 64)  wb_in[t]   = f2bf(in_w[t]);
  if (t < 64 * 128)  wb_out[t]  = f2bf(out_w[t]);
  if (t < 256 * 256) wb_proj[t] = f2bf(proj_w[t]);
  if (t < 128 * 16)  Aneg[t]    = -__expf(A_log[t]);
}

// ---------------- LayerNorm #1 + rearrange into parts (f32 + bf16) ----------------
__global__ void k_ln1(const float* x, const float* g, const float* b,
                      float* parts_f32, unsigned short* parts_bf16) {
  __shared__ float red[256];
  int row = blockIdx.x;           // n*L + l
  int n = row >> 12, l = row & 4095;
  int c = threadIdx.x;
  float v = x[((size_t)(n * CC + c)) * LL + l];
  red[c] = v; __syncthreads();
  for (int s = 128; s > 0; s >>= 1) { if (c < s) red[c] += red[c + s]; __syncthreads(); }
  float mu = red[0] * (1.f / CC); __syncthreads();
  float dv = v - mu;
  red[c] = dv * dv; __syncthreads();
  for (int s = 128; s > 0; s >>= 1) { if (c < s) red[c] += red[c + s]; __syncthreads(); }
  float var = red[0] * (1.f / CC);
  float yv = dv * rsqrtf(var + 1e-5f) * g[c] + b[c];
  int grp = c >> 6, cq = c & 63;
  size_t idx = ((size_t)((grp * BB + n) * LL + l)) * DM + cq;  // row m = npart*L + l
  parts_f32[idx] = yv;
  parts_bf16[idx] = f2bf(yv);
}

// ---------------- in_proj: xz[32768,256] = parts[32768,64] @ W[256,64]^T ----------------
// One wave per (M-tile, 4 N-tiles): A fragment reused 4x.
__global__ void k_gemm_in(const unsigned* A, const unsigned* W, float* xz) {
  int lane = threadIdx.x & 31;
  int warp = threadIdx.x >> 5;
  int wid = blockIdx.x * 8 + warp;       // 8192 waves = 2048(Mt) x 4(Ng)
  int mt = wid >> 2, ng = wid & 3;
  int m0 = mt * 16;
  v16bf a0 = load_frag_a(A, m0, 0, 64, lane);
  v16bf a1 = load_frag_a(A, m0, 32, 64, lane);
  int n = lane & 15, half = lane >> 4;
#pragma unroll
  for (int j = 0; j < 4; ++j) {
    int e0 = ng * 64 + j * 16;
    v16bf b0 = load_frag_b(W, e0, 0, 64, lane);
    v16bf b1 = load_frag_b(W, e0, 32, 64, lane);
    v8f c = {};
    c = __builtin_amdgcn_wmma_f32_16x16x32_bf16(false, a0, false, b0, (short)0, c, false, false);
    c = __builtin_amdgcn_wmma_f32_16x16x32_bf16(false, a1, false, b1, (short)0, c, false, false);
#pragma unroll
    for (int r = 0; r < 8; ++r)
      xz[(size_t)(m0 + r + 8 * half) * 256 + e0 + n] = c[r];
  }
}

// ---------------- depthwise causal conv(4) + SiLU over u half of xz ----------------
__global__ void k_conv(const float* xz, const float* cw, const float* cb, float* u) {
  int t = blockIdx.x * blockDim.x + threadIdx.x;   // (n*L + l)*128 + d
  int d = t & 127;
  int l = (t >> 7) & 4095;
  int n = t >> 19;
  float acc = cb[d];
#pragma unroll
  for (int j = 0; j < 4; ++j) {
    int ls = l - 3 + j;
    if (ls >= 0) acc += cw[d * 4 + j] * xz[((size_t)(n * LL + ls)) * 256 + d];
  }
  u[t] = silu_f(acc);
}

// ---------------- x_proj (N=36) + dt_proj + softplus (LDS-staged VALU) ----------------
__global__ void k_xproj(const float* u, const float* xw, const float* dtw,
                        const float* dtb, float* dtf, float* Bm, float* Cm) {
  __shared__ float su[128];
  __shared__ float sdt[4];
  int m = blockIdx.x;
  int t = threadIdx.x;
  su[t] = u[(size_t)m * 128 + t];
  __syncthreads();
  if (t < 36) {
    float acc = 0.f;
    const float* wr = xw + t * 128;
#pragma unroll 8
    for (int i = 0; i < 128; ++i) acc += wr[i] * su[i];
    if (t < 4) sdt[t] = acc;
    else if (t < 20) Bm[(size_t)m * DS + (t - 4)] = acc;
    else Cm[(size_t)m * DS + (t - 20)] = acc;
  }
  __syncthreads();
  float acc = dtb[t];
#pragma unroll
  for (int r = 0; r < 4; ++r) acc += dtw[t * 4 + r] * sdt[r];
  dtf[(size_t)m * 128 + t] = softplus_f(acc);
}

// ---------------- chunked scan pass 1: per-chunk (prod dA, h_end) ----------------
// unit = (n*128 + d)*16 + chunk ; 16 lanes per unit (k = state index).
// dt/u loaded once per 16-step block (lane k holds step k) and broadcast via shfl.
__global__ void k_scan1(const float* dtf, const float* u, const float* Bm,
                        const float* Aneg, float* prodA, float* hend) {
  int lane = threadIdx.x & 31;
  int unit = blockIdx.x * 16 + (threadIdx.x >> 4);
  int k = lane & 15;
  int srcbase = lane & 16;
  int chunk = unit & 15;
  int d = (unit >> 4) & 127;
  int n = unit >> 11;
  float a = Aneg[d * DS + k];
  float p = 1.f, h = 0.f;
  int base = n * LL + chunk * CLEN;
  for (int ib = 0; ib < CLEN / 16; ++ib) {
    int lb = base + ib * 16;
    float dtv = dtf[(size_t)(lb + k) * 128 + d];
    float uv  = u[(size_t)(lb + k) * 128 + d];
    __builtin_prefetch(&dtf[(size_t)(lb + 16 + k) * 128 + d], 0, 3);
    __builtin_prefetch(&u[(size_t)(lb + 16 + k) * 128 + d], 0, 3);
    __builtin_prefetch(&Bm[(size_t)(lb + 16 + k) * DS], 0, 3);
#pragma unroll
    for (int i = 0; i < 16; ++i) {
      float dt = __shfl(dtv, srcbase | i);
      float uu = __shfl(uv, srcbase | i);
      float bb = Bm[(size_t)(lb + i) * DS + k];
      float dA = __expf(dt * a);
      h = fmaf(dA, h, dt * bb * uu);
      p *= dA;
    }
  }
  prodA[(size_t)unit * DS + k] = p;
  hend[(size_t)unit * DS + k] = h;
}

// ---------------- pass 2: sequential combine over 16 chunks -> prefix states ----------------
__global__ void k_scan2(const float* prodA, const float* hend, float* hpref) {
  int nd = blockIdx.x * 16 + (threadIdx.x >> 4);   // (n*128 + d), 1024 total
  int k = threadIdx.x & 15;
  float h = 0.f;
  for (int c = 0; c < 16; ++c) {
    size_t idx = ((size_t)nd * 16 + c) * DS + k;
    hpref[idx] = h;
    h = prodA[idx] * h + hend[idx];
  }
}

// ---------------- pass 3: re-scan with prefix, y=<h,C>+D*u, fused gate y*silu(z)->bf16 ----------------
__global__ void k_scan3(const float* dtf, const float* u, const float* Bm, const float* Cm,
                        const float* xz, const float* Aneg, const float* Dp,
                        const float* hpref, unsigned short* yzb) {
  int lane = threadIdx.x & 31;
  int unit = blockIdx.x * 16 + (threadIdx.x >> 4);
  int k = lane & 15;
  int srcbase = lane & 16;
  int chunk = unit & 15;
  int d = (unit >> 4) & 127;
  int n = unit >> 11;
  float a = Aneg[d * DS + k];
  float dd = Dp[d];
  float h = hpref[(size_t)unit * DS + k];
  int base = n * LL + chunk * CLEN;
  for (int ib = 0; ib < CLEN / 16; ++ib) {
    int lb = base + ib * 16;
    float dtv = dtf[(size_t)(lb + k) * 128 + d];
    float uv  = u[(size_t)(lb + k) * 128 + d];
    float zv  = xz[(size_t)(lb + k) * 256 + 128 + d];
    __builtin_prefetch(&dtf[(size_t)(lb + 16 + k) * 128 + d], 0, 3);
    __builtin_prefetch(&u[(size_t)(lb + 16 + k) * 128 + d], 0, 3);
    __builtin_prefetch(&Bm[(size_t)(lb + 16 + k) * DS], 0, 3);
    __builtin_prefetch(&Cm[(size_t)(lb + 16 + k) * DS], 0, 3);
#pragma unroll
    for (int i = 0; i < 16; ++i) {
      float dt = __shfl(dtv, srcbase | i);
      float uu = __shfl(uv, srcbase | i);
      float zz = __shfl(zv, srcbase | i);
      float bb = Bm[(size_t)(lb + i) * DS + k];
      float cc = Cm[(size_t)(lb + i) * DS + k];
      float dA = __expf(dt * a);
      h = fmaf(dA, h, dt * bb * uu);
      float s = h * cc;
      s += __shfl_xor(s, 1);
      s += __shfl_xor(s, 2);
      s += __shfl_xor(s, 4);
      s += __shfl_xor(s, 8);        // stays within the 16-lane unit
      if (k == 0) {
        float yv = s + dd * uu;
        yzb[(size_t)(lb + i) * 128 + d] = f2bf(yv * silu_f(zz));
      }
    }
  }
}

// ---------------- out_proj (WMMA) + skip, scatter into x_mamba (B,L,256) ----------------
// One wave per (M-tile, 2 N-tiles): A reused 2x.
__global__ void k_gemm_out(const unsigned* A, const unsigned* W,
                           const float* parts_f32, const float* skip, float* xmamba) {
  int lane = threadIdx.x & 31;
  int warp = threadIdx.x >> 5;
  int wid = blockIdx.x * 8 + warp;      // 4096 waves = 2048(Mt) x 2(Ng)
  int mt = wid >> 1, ng = wid & 1;
  int m0 = mt * 16;
  int e0 = ng * 32;
  v8f c0 = {}, c1 = {};
#pragma unroll
  for (int kb = 0; kb < 128; kb += 32) {
    v16bf a  = load_frag_a(A, m0, kb, 128, lane);
    v16bf b0 = load_frag_b(W, e0,      kb, 128, lane);
    v16bf b1 = load_frag_b(W, e0 + 16, kb, 128, lane);
    c0 = __builtin_amdgcn_wmma_f32_16x16x32_bf16(false, a, false, b0, (short)0, c0, false, false);
    c1 = __builtin_amdgcn_wmma_f32_16x16x32_bf16(false, a, false, b1, (short)0, c1, false, false);
  }
  float sk = skip[0];
  int nn = lane & 15, half = lane >> 4;
#pragma unroll
  for (int j = 0; j < 2; ++j) {
    int e = e0 + j * 16;
#pragma unroll
    for (int r = 0; r < 8; ++r) {
      int m = m0 + r + 8 * half;          // npart*L + l
      int npart = m >> 12, l = m & 4095;
      int gidx = npart >> 1, nb = npart & 1;
      int cq = e + nn;
      float cv = (j == 0) ? c0[r] : c1[r];
      float v = cv + sk * parts_f32[(size_t)m * 64 + cq];
      xmamba[((size_t)(nb * LL + l)) * 256 + gidx * 64 + cq] = v;
    }
  }
}

// ---------------- LayerNorm #2 -> bf16 rows ----------------
__global__ void k_ln2(const float* xm, const float* g, const float* b, unsigned short* xmb) {
  __shared__ float red[256];
  int row = blockIdx.x;
  int c = threadIdx.x;
  float v = xm[(size_t)row * 256 + c];
  red[c] = v; __syncthreads();
  for (int s = 128; s > 0; s >>= 1) { if (c < s) red[c] += red[c + s]; __syncthreads(); }
  float mu = red[0] * (1.f / 256.f); __syncthreads();
  float dv = v - mu;
  red[c] = dv * dv; __syncthreads();
  for (int s = 128; s > 0; s >>= 1) { if (c < s) red[c] += red[c + s]; __syncthreads(); }
  float var = red[0] * (1.f / 256.f);
  xmb[(size_t)row * 256 + c] = f2bf(dv * rsqrtf(var + 1e-5f) * g[c] + b[c]);
}

// ---------------- final proj (WMMA) + bias, transposed store (B,C,L) ----------------
// One wave per (M-tile, 2 N-tiles): A reused 2x.
__global__ void k_gemm_final(const unsigned* A, const unsigned* W, const float* bias, float* out) {
  int lane = threadIdx.x & 31;
  int warp = threadIdx.x >> 5;
  int wid = blockIdx.x * 8 + warp;      // 4096 waves = 512(Mt) x 8(Ng)
  int mt = wid >> 3, ng = wid & 7;
  int m0 = mt * 16;
  int e0 = ng * 32;
  v8f c0 = {}, c1 = {};
#pragma unroll
  for (int kb = 0; kb < 256; kb += 32) {
    v16bf a  = load_frag_a(A, m0, kb, 256, lane);
    v16bf b0 = load_frag_b(W, e0,      kb, 256, lane);
    v16bf b1 = load_frag_b(W, e0 + 16, kb, 256, lane);
    c0 = __builtin_amdgcn_wmma_f32_16x16x32_bf16(false, a, false, b0, (short)0, c0, false, false);
    c1 = __builtin_amdgcn_wmma_f32_16x16x32_bf16(false, a, false, b1, (short)0, c1, false, false);
  }
  int nn = lane & 15, half = lane >> 4;
#pragma unroll
  for (int j = 0; j < 2; ++j) {
    int e = e0 + j * 16;
#pragma unroll
    for (int r = 0; r < 8; ++r) {
      int m = m0 + r + 8 * half;          // nb*L + l
      int nb = m >> 12, l = m & 4095;
      int o = e + nn;
      float cv = (j == 0) ? c0[r] : c1[r];
      out[((size_t)(nb * 256 + o)) * LL + l] = cv + bias[o];
    }
  }
}

extern "C" void kernel_launch(void* const* d_in, const int* in_sizes, int n_in,
                              void* d_out, int out_size, void* d_ws, size_t ws_size,
                              hipStream_t stream) {
  (void)in_sizes; (void)n_in; (void)out_size; (void)ws_size;
  const float* x       = (const float*)d_in[0];
  const float* ln_g    = (const float*)d_in[1];
  const float* ln_b    = (const float*)d_in[2];
  const float* skip    = (const float*)d_in[3];
  const float* in_w    = (const float*)d_in[4];
  const float* conv_w  = (const float*)d_in[5];
  const float* conv_b  = (const float*)d_in[6];
  const float* xproj_w = (const float*)d_in[7];
  const float* dtw     = (const float*)d_in[8];
  const float* dtb     = (const float*)d_in[9];
  const float* A_log   = (const float*)d_in[10];
  const float* Dp      = (const float*)d_in[11];
  const float* out_w   = (const float*)d_in[12];
  const float* proj_w  = (const float*)d_in[13];
  const float* proj_b  = (const float*)d_in[14];
  float* out = (float*)d_out;

  char* ws = (char*)d_ws;
  size_t off = 0;
  auto alloc = [&](size_t bytes) { void* p = ws + off; off += (bytes + 255) & ~(size_t)255; return p; };

  float*          parts_f32  = (float*)alloc((size_t)MROWS * 64 * 4);          // 8 MB
  unsigned short* parts_bf16 = (unsigned short*)alloc((size_t)MROWS * 64 * 2); // 4 MB
  float*          xz         = (float*)alloc((size_t)MROWS * 256 * 4);         // 32 MB
  float*          u_f32      = (float*)alloc((size_t)MROWS * 128 * 4);         // 16 MB
  float*          dtf        = (float*)alloc((size_t)MROWS * 128 * 4);         // 16 MB
  float*          Bm         = (float*)alloc((size_t)MROWS * 16 * 4);          // 2 MB
  float*          Cm         = (float*)alloc((size_t)MROWS * 16 * 4);          // 2 MB
  float*          prodA      = (float*)alloc((size_t)NP * 128 * 16 * 16 * 4);  // 1 MB
  float*          hend       = (float*)alloc((size_t)NP * 128 * 16 * 16 * 4);  // 1 MB
  float*          hpref      = (float*)alloc((size_t)NP * 128 * 16 * 16 * 4);  // 1 MB
  unsigned short* yzb        = (unsigned short*)alloc((size_t)MROWS * 128 * 2);// 8 MB
  float*          xmamba     = (float*)alloc((size_t)BB * LL * 256 * 4);       // 8 MB
  unsigned short* xmb        = (unsigned short*)alloc((size_t)BB * LL * 256 * 2); // 4 MB
  unsigned short* wb_in      = (unsigned short*)alloc(256 * 64 * 2);
  unsigned short* wb_out     = (unsigned short*)alloc(64 * 128 * 2);
  unsigned short* wb_proj    = (unsigned short*)alloc(256 * 256 * 2);
  float*          Aneg       = (float*)alloc(128 * 16 * 4);

  k_repack<<<256, 256, 0, stream>>>(in_w, out_w, proj_w, A_log, wb_in, wb_out, wb_proj, Aneg);
  k_ln1<<<BB * LL, 256, 0, stream>>>(x, ln_g, ln_b, parts_f32, parts_bf16);
  k_gemm_in<<<1024, 256, 0, stream>>>((const unsigned*)parts_bf16, (const unsigned*)wb_in, xz);
  k_conv<<<(MROWS * 128) / 256, 256, 0, stream>>>(xz, conv_w, conv_b, u_f32);
  k_xproj<<<MROWS, 128, 0, stream>>>(u_f32, xproj_w, dtw, dtb, dtf, Bm, Cm);
  k_scan1<<<1024, 256, 0, stream>>>(dtf, u_f32, Bm, Aneg, prodA, hend);
  k_scan2<<<64, 256, 0, stream>>>(prodA, hend, hpref);
  k_scan3<<<1024, 256, 0, stream>>>(dtf, u_f32, Bm, Cm, xz, Aneg, Dp, hpref, yzb);
  k_gemm_out<<<512, 256, 0, stream>>>((const unsigned*)yzb, (const unsigned*)wb_out, parts_f32, skip, xmamba);
  k_ln2<<<BB * LL, 256, 0, stream>>>(xmamba, ln_g, ln_b, xmb);
  k_gemm_final<<<512, 256, 0, stream>>>((const unsigned*)xmb, (const unsigned*)wb_proj, proj_b, out);
}
